// SelfAttentionBlock_22050362098273
// MI455X (gfx1250) — matrix operand order repaired and checked by reading, not compiled, not run on hardware
//
#include <hip/hip_runtime.h>
#include <hip/hip_bf16.h>

// ---------------------------------------------------------------------------
// Problem constants (fixed by setup_inputs): B=16, C=256, H=W=64
// ---------------------------------------------------------------------------
#define BATCH 16
#define CIN   256
#define C2    128
#define HH    64
#define WW    64
#define NPIX  (HH * WW)        // 4096
#define MPIX  ((HH/2)*(WW/2))  // 1024

typedef __attribute__((ext_vector_type(16))) __bf16 v16bf;
typedef __attribute__((ext_vector_type(8)))  float  v8f;

union Frag16 {
  v16bf v;
  unsigned int u[8];
};

// float -> bf16 bits, round-to-nearest-even
static __device__ __forceinline__ unsigned short f2bf(float f) {
  unsigned int u = __float_as_uint(f);
  u += 0x7FFFu + ((u >> 16) & 1u);
  return (unsigned short)(u >> 16);
}
static __device__ __forceinline__ float bf2f(unsigned short b) {
  return __uint_as_float(((unsigned int)b) << 16);
}

// A fragment: 16x32 bf16, M x K, src points at (row0, k0) of row-major [_, ld]
// ISA 7.12.2: lanes0-15 row M=lane; VGPR0-3: K = half*8 + 2p; VGPR4-7: K = 16 + half*8 + 2p
static __device__ __forceinline__ v16bf load_frag_a(const unsigned short* src, int ld) {
  const int lane = threadIdx.x & 31;
  const int row  = lane & 15;
  const int half = lane >> 4;
  const unsigned short* p = src + (size_t)row * ld;
  Frag16 f;
#pragma unroll
  for (int i = 0; i < 4; ++i)
    f.u[i] = *(const unsigned int*)(p + half * 8 + 2 * i);
#pragma unroll
  for (int i = 0; i < 4; ++i)
    f.u[4 + i] = *(const unsigned int*)(p + 16 + half * 8 + 2 * i);
  return f.v;
}

// B fragment: 32x16 bf16, K x N. Element (k, col) lives at src[col*ldc + k]
// (K-major per column). Lane = col (mod 16); VGPR0-3: K = half*8 + 2p; VGPR4-7: K = 16 + half*8 + 2p.
static __device__ __forceinline__ v16bf load_frag_b(const unsigned short* src, int ldc) {
  const int lane = threadIdx.x & 31;
  const int col  = lane & 15;
  const int half = lane >> 4;
  const unsigned short* p = src + (size_t)col * ldc;
  Frag16 f;
#pragma unroll
  for (int i = 0; i < 4; ++i)
    f.u[i] = *(const unsigned int*)(p + half * 8 + 2 * i);
#pragma unroll
  for (int i = 0; i < 4; ++i)
    f.u[4 + i] = *(const unsigned int*)(p + 16 + half * 8 + 2 * i);
  return f.v;
}

static __device__ __forceinline__ v8f wmma_bf16(v16bf a, v16bf b, v8f c) {
  return __builtin_amdgcn_wmma_f32_16x16x32_bf16(false, a, false, b, (short)0, c,
                                                 false, false);
}

// ---------------------------------------------------------------------------
// Async global -> LDS staging of a 128-row x 16-col bf16 slab (4 KB; columns 0-15 of each 32-col LDS row).
// One 16-byte chunk per thread (256 threads). Uses the gfx1250 ASYNC DMA
// path (GLOBAL_LOAD_ASYNC_TO_LDS_B128, ASYNCcnt) when available.
// Builtin signature (from clang diagnostic): param0 = int4 __device__*(AS1),
// param1 = LDS pointer, then i32 offset, i32 cpol.
// ---------------------------------------------------------------------------
#if defined(__has_builtin)
#if __has_builtin(__builtin_amdgcn_global_load_async_to_lds_b128)
#define HAVE_ASYNC_LDS 1
#endif
#endif
#ifndef HAVE_ASYNC_LDS
#define HAVE_ASYNC_LDS 0
#endif

typedef int v4i_t __attribute__((vector_size(4 * sizeof(int))));

static __device__ __forceinline__ void stage_slab_128x32(
    const unsigned short* __restrict__ gsrc, int srcld,
    unsigned short* lds, int tid) {
  const int o = tid >> 1;
  const int q = (tid & 1) * 8;  // 8 ushorts = 16 bytes
#if HAVE_ASYNC_LDS
  __builtin_amdgcn_global_load_async_to_lds_b128(
      (__attribute__((address_space(1))) v4i_t*)(gsrc + (size_t)o * srcld + q),
      (__attribute__((address_space(3))) v4i_t*)(lds + o * 32 + q), 0, 0);
#else
  uint4 v = *(const uint4*)(gsrc + (size_t)o * srcld + q);
  *(uint4*)(lds + o * 32 + q) = v;
#endif
}

static __device__ __forceinline__ void stage_wait() {
#if HAVE_ASYNC_LDS
#if __has_builtin(__builtin_amdgcn_s_wait_asynccnt)
  __builtin_amdgcn_s_wait_asynccnt(0);
#else
  asm volatile("s_wait_asynccnt 0" ::: "memory");
#endif
#endif
  __syncthreads();
}

// ---------------------------------------------------------------------------
// K0: f32 -> bf16 weight conversion
// ---------------------------------------------------------------------------
__global__ void cvt_bf16_kernel(const float* __restrict__ src,
                                unsigned short* __restrict__ dst, int n) {
  int i = blockIdx.x * 256 + threadIdx.x;
  if (i < n) dst[i] = f2bf(src[i]);
}

// ---------------------------------------------------------------------------
// K1: x [B, C, N] f32  ->  xb16 [B, N, C] bf16   (A-fragment ready)
// tile: 32 c x 64 n per block
// ---------------------------------------------------------------------------
__global__ __launch_bounds__(256) void transpose_x_kernel(
    const float* __restrict__ x, unsigned short* __restrict__ xb) {
  __shared__ float tile[32][65];
  const int nt = blockIdx.x;  // 64 tiles of 64 n
  const int ct = blockIdx.y;  // 8 tiles of 32 c
  const int b  = blockIdx.z;
  const int tid = threadIdx.x;

  for (int i = tid; i < 32 * 64; i += 256) {
    int c = i >> 6, n = i & 63;
    tile[c][n] = x[((size_t)b * CIN + ct * 32 + c) * NPIX + nt * 64 + n];
  }
  __syncthreads();
  for (int i = tid; i < 64 * 16; i += 256) {
    int n = i >> 4, cp = (i & 15) * 2;
    unsigned int v = (unsigned int)f2bf(tile[cp][n]) |
                     ((unsigned int)f2bf(tile[cp + 1][n]) << 16);
    *(unsigned int*)&xb[((size_t)b * NPIX + nt * 64 + n) * CIN + ct * 32 + cp] = v;
  }
}

// ---------------------------------------------------------------------------
// K2: fused theta/phi/g projection GEMM + bias + 2x2 maxpool
// block = 256 threads (8 waves); tile = 128 n (two h-rows) x 128 outputs
// Weight K-slabs staged in LDS (async DMA) and shared by all 8 waves.
// theta -> [B, N, C2] bf16 ; phi -> [B, M, C2] bf16 ; g -> [B, C2, M] bf16
// ---------------------------------------------------------------------------
#define STG_LD 132
__global__ __launch_bounds__(256) void proj_pool_kernel(
    const unsigned short* __restrict__ xb,
    const unsigned short* __restrict__ wt,
    const unsigned short* __restrict__ wp,
    const unsigned short* __restrict__ wg,
    const float* __restrict__ bt, const float* __restrict__ bp,
    const float* __restrict__ bg,
    unsigned short* __restrict__ theta_o,
    unsigned short* __restrict__ phi_o,
    unsigned short* __restrict__ g_o) {
  __shared__ unsigned short stage[128 * STG_LD];  // 33 KB result staging
  __shared__ unsigned short wslab[128 * 32];      // 8 KB weight K-slab

  const int hp  = blockIdx.x;   // 32 h-row pairs
  const int b   = blockIdx.y;
  const int tid = threadIdx.x;
  const int wid  = tid >> 5;
  const int lane = tid & 31;
  const int coll = lane & 15;
  const int half = lane >> 4;
  const int nbase = hp * 128;
  const int n0 = nbase + wid * 16;

  const unsigned short* xrow = xb + ((size_t)b * NPIX + n0) * CIN;

  // x A-fragments for this wave's 16 rows, K = 256 -> 8 frags, reused for all
  // three projections.
  v16bf aX[8];
#pragma unroll
  for (int kc = 0; kc < 8; ++kc) aX[kc] = load_frag_a(xrow + kc * 32, CIN);

  for (int t = 0; t < 3; ++t) {
    const unsigned short* wsel = (t == 0) ? wt : (t == 1) ? wp : wg;
    const float*          bsel = (t == 0) ? bt : (t == 1) ? bp : bg;

    v8f acc[8];
#pragma unroll
    for (int o = 0; o < 8; ++o) acc[o] = (v8f){};

    for (int kc = 0; kc < 8; ++kc) {           // K = 256 = 8 x 32
      __syncthreads();  // previous slab fully consumed
      stage_slab_128x32(wsel + kc * 32, CIN, wslab, tid);
      stage_wait();
#pragma unroll
      for (int o = 0; o < 8; ++o) {
        v16bf bw = load_frag_b(wslab + (o * 16) * 32, 32);
        acc[o] = wmma_bf16(aX[kc], bw, acc[o]);
      }
    }
    // bias + stage to LDS in [n_local][o] layout
#pragma unroll
    for (int o = 0; o < 8; ++o) {
      float bias = bsel[o * 16 + coll];
#pragma unroll
      for (int r = 0; r < 8; ++r) {
        float v = acc[o][r] + bias;
        stage[(wid * 16 + r + 8 * half) * STG_LD + o * 16 + coll] = f2bf(v);
      }
    }
    __syncthreads();

    if (t == 0) {  // theta: straight copy, [n, c2] row-major (pairs in c2)
      for (int i = tid; i < 128 * 64; i += 256) {
        int nl = i >> 6, oc = (i & 63) * 2;
        unsigned int v = *(unsigned int*)&stage[nl * STG_LD + oc];
        *(unsigned int*)&theta_o[((size_t)b * NPIX + nbase + nl) * C2 + oc] = v;
      }
    } else {       // phi / g: 2x2 maxpool over (h-row pair, w pair)
      for (int i = tid; i < 32 * 128; i += 256) {
        int wp2 = i >> 7, o = i & 127;
        float v0 = bf2f(stage[(2 * wp2)     * STG_LD + o]);
        float v1 = bf2f(stage[(2 * wp2 + 1) * STG_LD + o]);
        float v2 = bf2f(stage[(64 + 2 * wp2)     * STG_LD + o]);
        float v3 = bf2f(stage[(64 + 2 * wp2 + 1) * STG_LD + o]);
        float m4 = fmaxf(fmaxf(v0, v1), fmaxf(v2, v3));
        int m = hp * 32 + wp2;
        if (t == 1)  // phi: [m, c2] (K-pairs contiguous in c2 for QK^T B-frag)
          phi_o[(size_t)b * MPIX * C2 + (size_t)m * C2 + o] = f2bf(m4);
        else         // g: [c2, m] (K-pairs contiguous in m for PV B-frag)
          g_o[(size_t)b * C2 * MPIX + (size_t)o * MPIX + m] = f2bf(m4);
      }
    }
    __syncthreads();
  }
}

// ---------------------------------------------------------------------------
// K3: attention.  One wave per 16-row tile of theta.
//   S = theta(16x128) x phi^T(128x1024) -> LDS strip (bf16)
//   softmax rows over M=1024
//   out2 = P(16x1024) x g(1024x128)
// phi/g fragment loads partition the matrices exactly (no over-fetch) and the
// 512 KB/batch working set lives in L2 across the 256 row-tiles per batch.
// ---------------------------------------------------------------------------
__global__ __launch_bounds__(32) void attention_kernel(
    const unsigned short* __restrict__ theta,
    const unsigned short* __restrict__ phi,
    const unsigned short* __restrict__ g,
    unsigned short* __restrict__ out2) {
  __shared__ unsigned short P[16 * MPIX];  // 32 KB logits/probs strip

  const int tile = blockIdx.x;   // 256 tiles of 16 rows
  const int b    = blockIdx.y;
  const int lane = threadIdx.x & 31;
  const int coll = lane & 15;
  const int half = lane >> 4;
  const int n0   = tile * 16;

  const unsigned short* th = theta + ((size_t)b * NPIX + n0) * C2;
  const unsigned short* ph = phi + (size_t)b * MPIX * C2;
  const unsigned short* gb = g + (size_t)b * C2 * MPIX;

  // theta A-fragments, K = 128 -> 4 frags, held in registers
  v16bf aTh[4];
#pragma unroll
  for (int kc = 0; kc < 4; ++kc) aTh[kc] = load_frag_a(th + kc * 32, C2);

  // ---- pass 1: logits -> LDS ----
  for (int mt = 0; mt < MPIX / 16; ++mt) {
    v8f s = (v8f){};
#pragma unroll
    for (int kc = 0; kc < 4; ++kc) {
      v16bf bf = load_frag_b(ph + (size_t)(mt * 16) * C2 + kc * 32, C2);
      s = wmma_bf16(aTh[kc], bf, s);
    }
#pragma unroll
    for (int r = 0; r < 8; ++r)
      P[(r + 8 * half) * MPIX + mt * 16 + coll] = f2bf(s[r]);
  }
  __syncthreads();

  // ---- pass 2: row softmax (thread -> row lane&15, half handles 512 cols) ----
  {
    const int row = coll;
    unsigned short* prow = &P[row * MPIX + half * 512];
    float mx = -3.4e38f;
    for (int j = 0; j < 512; ++j) mx = fmaxf(mx, bf2f(prow[j]));
    mx = fmaxf(mx, __shfl_xor(mx, 16));
    float sum = 0.f;
    for (int j = 0; j < 512; ++j) sum += __expf(bf2f(prow[j]) - mx);
    sum += __shfl_xor(sum, 16);
    float inv = 1.f / sum;
    for (int j = 0; j < 512; ++j)
      prow[j] = f2bf(__expf(bf2f(prow[j]) - mx) * inv);
  }
  __syncthreads();

  // ---- pass 3: out2 = P x g ----
  v8f acc[8];
#pragma unroll
  for (int o = 0; o < 8; ++o) acc[o] = (v8f){};
  for (int kc = 0; kc < MPIX / 32; ++kc) {
    v16bf a = load_frag_a(&P[kc * 32], MPIX);
#pragma unroll
    for (int o = 0; o < 8; ++o) {
      v16bf bf = load_frag_b(gb + (size_t)(o * 16) * MPIX + kc * 32, MPIX);
      acc[o] = wmma_bf16(a, bf, acc[o]);
    }
  }
#pragma unroll
  for (int o = 0; o < 8; ++o)
#pragma unroll
    for (int r = 0; r < 8; ++r)
      out2[((size_t)b * NPIX + n0 + r + 8 * half) * C2 + o * 16 + coll] =
          f2bf(acc[o][r]);
}

// ---------------------------------------------------------------------------
// K4: out = x + out2 x w_out^T + b_out   (f32 output [B, C, N])
// block = 256 (8 waves), tile = 128 n x 128 co, grid.y picks co half.
// w_out K-slabs staged in LDS and shared by all 8 waves.
// ---------------------------------------------------------------------------
__global__ __launch_bounds__(256) void output_kernel(
    const unsigned short* __restrict__ out2,
    const unsigned short* __restrict__ wout,
    const float* __restrict__ bout, const float* __restrict__ x,
    float* __restrict__ out) {
  __shared__ unsigned short wslab[128 * 32];  // 8 KB weight K-slab

  const int nt = blockIdx.x;     // 32 tiles of 128 n
  const int ch = blockIdx.y;     // 2 halves of C=256
  const int b  = blockIdx.z;
  const int tid  = threadIdx.x;
  const int wid  = tid >> 5;
  const int lane = tid & 31;
  const int coll = lane & 15;
  const int half = lane >> 4;
  const int n0 = nt * 128 + wid * 16;
  const int c0 = ch * 128;

  const unsigned short* arow = out2 + ((size_t)b * NPIX + n0) * C2;

  // out2 A-fragments, K = C2 = 128 -> 4 frags
  v16bf aO[4];
#pragma unroll
  for (int kc = 0; kc < 4; ++kc) aO[kc] = load_frag_a(arow + kc * 32, C2);

  v8f acc[8];
#pragma unroll
  for (int o = 0; o < 8; ++o) acc[o] = (v8f){};

  for (int kc = 0; kc < 4; ++kc) {
    __syncthreads();
    stage_slab_128x32(wout + (size_t)c0 * C2 + kc * 32, C2, wslab, tid);
    stage_wait();
#pragma unroll
    for (int o = 0; o < 8; ++o) {
      v16bf bw = load_frag_b(wslab + (o * 16) * 32, 32);
      acc[o] = wmma_bf16(aO[kc], bw, acc[o]);
    }
  }

#pragma unroll
  for (int o = 0; o < 8; ++o) {
    int co = c0 + o * 16 + coll;
    float bias = bout[co];
#pragma unroll
    for (int r = 0; r < 8; ++r) {
      int n = n0 + r + 8 * half;
      size_t idx = ((size_t)b * CIN + co) * NPIX + n;
      out[idx] = x[idx] + acc[o][r] + bias;
    }
  }
}

// ---------------------------------------------------------------------------
// launch
// ---------------------------------------------------------------------------
extern "C" void kernel_launch(void* const* d_in, const int* in_sizes, int n_in,
                              void* d_out, int out_size, void* d_ws,
                              size_t ws_size, hipStream_t stream) {
  const float* x       = (const float*)d_in[0];
  const float* w_theta = (const float*)d_in[1];
  const float* b_theta = (const float*)d_in[2];
  const float* w_phi   = (const float*)d_in[3];
  const float* b_phi   = (const float*)d_in[4];
  const float* w_g     = (const float*)d_in[5];
  const float* b_g     = (const float*)d_in[6];
  const float* w_out   = (const float*)d_in[7];
  const float* b_out   = (const float*)d_in[8];

  char* ws = (char*)d_ws;
  size_t off = 0;
  unsigned short* xb16 = (unsigned short*)(ws + off); off += (size_t)BATCH * NPIX * CIN * 2;
  unsigned short* th16 = (unsigned short*)(ws + off); off += (size_t)BATCH * NPIX * C2 * 2;
  unsigned short* ph16 = (unsigned short*)(ws + off); off += (size_t)BATCH * MPIX * C2 * 2;
  unsigned short* g16  = (unsigned short*)(ws + off); off += (size_t)BATCH * C2 * MPIX * 2;
  unsigned short* o216 = (unsigned short*)(ws + off); off += (size_t)BATCH * NPIX * C2 * 2;
  unsigned short* wt16 = (unsigned short*)(ws + off); off += (size_t)C2 * CIN * 2;
  unsigned short* wp16 = (unsigned short*)(ws + off); off += (size_t)C2 * CIN * 2;
  unsigned short* wg16 = (unsigned short*)(ws + off); off += (size_t)C2 * CIN * 2;
  unsigned short* wo16 = (unsigned short*)(ws + off); off += (size_t)CIN * C2 * 2;

  // K0: weight conversion
  cvt_bf16_kernel<<<(C2 * CIN + 255) / 256, 256, 0, stream>>>(w_theta, wt16, C2 * CIN);
  cvt_bf16_kernel<<<(C2 * CIN + 255) / 256, 256, 0, stream>>>(w_phi, wp16, C2 * CIN);
  cvt_bf16_kernel<<<(C2 * CIN + 255) / 256, 256, 0, stream>>>(w_g, wg16, C2 * CIN);
  cvt_bf16_kernel<<<(CIN * C2 + 255) / 256, 256, 0, stream>>>(w_out, wo16, CIN * C2);

  // K1: x transpose+convert
  transpose_x_kernel<<<dim3(NPIX / 64, CIN / 32, BATCH), 256, 0, stream>>>(x, xb16);

  // K2: projections + pool (weights staged via async LDS DMA)
  proj_pool_kernel<<<dim3(HH / 2, BATCH), 256, 0, stream>>>(
      xb16, wt16, wp16, wg16, b_theta, b_phi, b_g, th16, ph16, g16);

  // K3: fused attention (softmax never touches HBM)
  attention_kernel<<<dim3(NPIX / 16, BATCH), 32, 0, stream>>>(th16, ph16, g16, o216);

  // K4: output projection + bias + residual
  output_kernel<<<dim3(NPIX / 128, 2, BATCH), 256, 0, stream>>>(
      o216, wo16, b_out, x, (float*)d_out);
}
